// RNN_50_100_18545668784386
// MI455X (gfx1250) — matrix operand order repaired and checked by reading, not compiled, not run on hardware
//
#include <hip/hip_runtime.h>

typedef __bf16 bf16;
typedef bf16  v16bf __attribute__((ext_vector_type(16)));
typedef float v8f   __attribute__((ext_vector_type(8)));
typedef unsigned int v4u __attribute__((ext_vector_type(4)));   // POD 16-byte vector

// global-address-space element types (force global_load/global_store codegen)
typedef __attribute__((address_space(1))) const v4u    g_cv4u;
typedef __attribute__((address_space(1))) const v8f    g_cv8f;
typedef __attribute__((address_space(1)))       v8f    g_v8f;
typedef __attribute__((address_space(1))) const bf16   g_cbf16;
typedef __attribute__((address_space(1)))       float  g_float;

#define H        256
#define G4       1024      // 4*H
#define Bsz      8192
#define Tenc     50
#define Din      50
#define NDEC     100
#define NY       100
#define MROWS    32        // batch rows per workgroup
#define NWG      (Bsz / MROWS)   // 256 workgroups
#define NT_G     (G4 / 16)       // 64 N-tiles of the gate matrix
#define NT_Y     7               // ceil(100/16)

// ---- workspace layout (element offsets for bf16 region, byte offsets after) ----
#define WENC_ELE  0                         // 320*1024 bf16
#define WDEC_ELE  (WENC_ELE + 320*1024)     // 256*1024 bf16
#define WCIN_ELE  (WDEC_ELE + 256*1024)     // 256*1024 bf16
#define WLIN_ELE  (WCIN_ELE + 256*1024)     // 256*112  bf16
#define WS_BF16_END (WLIN_ELE + 256*112)    // 880640 bf16 = 1761280 B
#define BIAS_BYTE (WS_BF16_END * 2)         // 1024 f32
#define IPROJ_BYTE (BIAS_BYTE + 1024*4)     // 8192*1024 f32 (32B aligned)

// Make a (uniform) pointer opaque to LICM/CSE so weight fragments are
// re-loaded from L2 every recurrent step instead of being hoisted into
// ~640 VGPRs and spilled to scratch. Pointer keeps its addrspace(1) type.
#define OPAQUE_S(p) asm volatile("" : "+s"(p))

__device__ __forceinline__ bf16 f2bf(float f) { return (bf16)f; }

// packed-fragment offset (in bf16 elements) for B-matrix element (k, n)
__device__ __forceinline__ int pack_off(int k, int n, int nTiles) {
    int kt = k >> 5, kk = k & 31;
    int nt = n >> 4, nn = n & 15;
    int lane = nn + ((kk >> 4) << 4);
    int sub  = kk & 15;
    return ((kt * nTiles + nt) << 9) + lane * 16 + sub;
}

// ---------------- weight packing kernels ----------------
__global__ void pack_enc(const float* __restrict__ W_ih, const float* __restrict__ W_hh,
                         const float* __restrict__ b_ih, const float* __restrict__ b_hh,
                         bf16* __restrict__ dst) {
    int e = blockIdx.x * blockDim.x + threadIdx.x;     // 320*1024 elements
    if (e >= 320 * G4) return;
    int k = e >> 10, n = e & (G4 - 1);
    float v;
    if (k < Din)            v = W_ih[n * Din + k];
    else if (k == Din)      v = b_ih[n] + b_hh[n];     // bias row (x col 50 == 1)
    else if (k < 64)        v = 0.f;
    else                    v = W_hh[n * H + (k - 64)];
    dst[pack_off(k, n, NT_G)] = f2bf(v);
}

__global__ void pack_h256(const float* __restrict__ src, bf16* __restrict__ dst) {
    int e = blockIdx.x * blockDim.x + threadIdx.x;     // 256*1024 elements
    if (e >= H * G4) return;
    int k = e >> 10, n = e & (G4 - 1);
    dst[pack_off(k, n, NT_G)] = f2bf(src[n * H + k]);
}

__global__ void pack_lin(const float* __restrict__ W_lin, bf16* __restrict__ dst) {
    int e = blockIdx.x * blockDim.x + threadIdx.x;     // 256*112 elements
    if (e >= H * NT_Y * 16) return;
    int k = e / (NT_Y * 16), n = e % (NT_Y * 16);
    float v = (n < NY) ? W_lin[n * H + k] : 0.f;
    dst[pack_off(k, n, NT_Y)] = f2bf(v);
}

__global__ void pack_bias(const float* __restrict__ bc_ih, const float* __restrict__ bc_hh,
                          float* __restrict__ dst) {
    int e = blockIdx.x * blockDim.x + threadIdx.x;
    if (e < G4) dst[e] = bc_ih[e] + bc_hh[e];
}

// ---------------- fragment loaders ----------------
__device__ __forceinline__ v16bf ld_a_lds(const bf16* buf, int mbase, int k0, int ld, int lane) {
    int m = mbase + (lane & 15);
    int k = k0 + ((lane >> 4) << 4);
    union { v4u q[2]; v16bf v; } u;
    u.q[0] = *(const v4u*)(buf + m * ld + k);          // K = k .. k+7
    u.q[1] = *(const v4u*)(buf + m * ld + k + 8);      // K = k+8 .. k+15
    return u.v;
}

__device__ __forceinline__ v16bf ld_b_pk(const g_cbf16* W, int block, int lane) {
    const g_cv4u* p = (const g_cv4u*)(W + ((size_t)block << 9) + lane * 16);
    union { v4u q[2]; v16bf v; } u;
    u.q[0] = p[0]; u.q[1] = p[1];
    return u.v;
}

__device__ __forceinline__ v8f wmma_bf16(v16bf a, v16bf b, v8f c) {
    return __builtin_amdgcn_wmma_f32_16x16x32_bf16(false, a, false, b, (short)0, c, false, false);
}

__device__ __forceinline__ float sigm(float x) { return 1.f / (1.f + __expf(-x)); }

// gate accumulation over packed weight W (nTiles = 64), K tiles from LDS source(s)
// acc indexed [gate][j][mt]
__device__ __forceinline__ void gates_mma(v8f acc[4][2][2], const g_cbf16* W,
                                          const bf16* hbuf, const bf16* xbuf,
                                          int kTiles, int xTiles, int wave, int lane) {
    OPAQUE_S(W);   // defeat hoisting of B fragments out of the time loop
    for (int kt = 0; kt < kTiles; ++kt) {
        v16bf a0, a1;
        if (kt < xTiles) {
            a0 = ld_a_lds(xbuf, 0, kt * 32, 64, lane);
            a1 = ld_a_lds(xbuf, 16, kt * 32, 64, lane);
        } else {
            a0 = ld_a_lds(hbuf, 0, (kt - xTiles) * 32, H, lane);
            a1 = ld_a_lds(hbuf, 16, (kt - xTiles) * 32, H, lane);
        }
        // stage all 8 weight fragments of this K-tile -> 8 loads in flight,
        // incremental s_wait_loadcnt before each WMMA pair
        v16bf b[8];
#pragma unroll
        for (int gj = 0; gj < 8; ++gj) {
            int ntG = (gj >> 1) * 16 + wave * 2 + (gj & 1);
            b[gj] = ld_b_pk(W, kt * NT_G + ntG, lane);
        }
        if (kt + 1 < kTiles)   // next K-tile of the L2-resident weight slab
            __builtin_prefetch((const void*)(const bf16*)(W + (((size_t)(kt + 1) * NT_G + wave * 2) << 9)),
                               0, 3);
#pragma unroll
        for (int gj = 0; gj < 8; ++gj) {
            acc[gj >> 1][gj & 1][0] = wmma_bf16(a0, b[gj], acc[gj >> 1][gj & 1][0]);
            acc[gj >> 1][gj & 1][1] = wmma_bf16(a1, b[gj], acc[gj >> 1][gj & 1][1]);
        }
    }
}

// elementwise LSTM cell update; writes new h (bf16) into hbuf
__device__ __forceinline__ void lstm_update(v8f acc[4][2][2], v8f c[2][2],
                                            bf16* hbuf, int wave, int lane) {
#pragma unroll
    for (int j = 0; j < 2; ++j)
#pragma unroll
        for (int mt = 0; mt < 2; ++mt) {
            v8f hn;
#pragma unroll
            for (int e = 0; e < 8; ++e) {
                float ig = sigm(acc[0][j][mt][e]);
                float fg = sigm(acc[1][j][mt][e]);
                float gg = tanhf(acc[2][j][mt][e]);
                float og = sigm(acc[3][j][mt][e]);
                float cn = fg * c[j][mt][e] + ig * gg;
                c[j][mt][e] = cn;
                hn[e] = og * tanhf(cn);
            }
            int n  = wave * 32 + j * 16 + (lane & 15);
            int mb = mt * 16 + ((lane >> 4) << 3);
#pragma unroll
            for (int e = 0; e < 8; ++e)
                hbuf[(mb + e) * H + n] = f2bf(hn[e]);
        }
}

// ---------------- fused persistent LSTM kernel ----------------
__global__ __launch_bounds__(256, 1)
void lstm_fused(const float* __restrict__ x,
                const bf16*  __restrict__ Wenc,
                const bf16*  __restrict__ Wdec,
                const bf16*  __restrict__ Wcin,
                const bf16*  __restrict__ Wlin,
                const float* __restrict__ bias_dec,
                const float* __restrict__ b_lin,
                float* __restrict__ out,
                float* __restrict__ iproj) {
    __shared__ __align__(16) bf16 hbuf[MROWS * H];     // 16 KB
    __shared__ __align__(16) bf16 xbuf[MROWS * 64];    //  4 KB

    const int tid  = threadIdx.x;
    const int wave = tid >> 5;
    const int lane = tid & 31;
    const int m0   = blockIdx.x * MROWS;

    const g_cbf16* WencG = (const g_cbf16*)Wenc;
    const g_cbf16* WdecG = (const g_cbf16*)Wdec;
    const g_cbf16* WcinG = (const g_cbf16*)Wcin;
    const g_cbf16* WlinG = (const g_cbf16*)Wlin;

    for (int i = tid; i < MROWS * H; i += 256) hbuf[i] = f2bf(0.f);

    v8f c[2][2];
#pragma unroll
    for (int j = 0; j < 2; ++j)
#pragma unroll
        for (int mt = 0; mt < 2; ++mt) c[j][mt] = (v8f)0.f;

    // ---------------- encoder: 50 steps ----------------
    for (int t = 0; t < Tenc; ++t) {
        // stage x_t tile (bf16, padded to 64 cols, col 50 = 1.0 for bias row)
        for (int i = tid; i < MROWS * 64; i += 256) {
            int r = i >> 6, d = i & 63;
            float v = (d < Din) ? x[(size_t)(m0 + r) * (Tenc * Din) + t * Din + d]
                                : (d == Din ? 1.f : 0.f);
            xbuf[i] = f2bf(v);
        }
        __syncthreads();
        v8f acc[4][2][2] = {};
        gates_mma(acc, WencG, hbuf, xbuf, 10, 2, wave, lane);
        __syncthreads();
        lstm_update(acc, c, hbuf, wave, lane);
        __syncthreads();
    }

    // ---------------- inp_proj = h_T @ Wc_ih^T + bc (constant per decode step) ----------------
    g_float* ipBase = (g_float*)(iproj + (size_t)blockIdx.x * (MROWS * G4));
    {
        v8f acc[4][2][2] = {};
        gates_mma(acc, WcinG, hbuf, xbuf, 8, 0, wave, lane);
#pragma unroll
        for (int g = 0; g < 4; ++g)
#pragma unroll
            for (int j = 0; j < 2; ++j) {
                int n  = g * H + wave * 32 + j * 16 + (lane & 15);
                float bv = bias_dec[n];
#pragma unroll
                for (int mt = 0; mt < 2; ++mt) {
                    v8f a = acc[g][j][mt];
#pragma unroll
                    for (int e = 0; e < 8; ++e) a[e] += bv;
                    int ti = (g * 2 + j) * 2 + mt;
                    *(g_v8f*)(ipBase + ((size_t)(wave * 16 + ti) * 32 + lane) * 8) = a;
                }
            }
    }

    // ---------------- decoder: 100 steps ----------------
    for (int s = 0; s < NDEC; ++s) {
        __syncthreads();
        const g_float* ip = ipBase;
        OPAQUE_S(ip);                 // reload accumulator init from L2 each step
        v8f acc[4][2][2];
#pragma unroll
        for (int g = 0; g < 4; ++g)
#pragma unroll
            for (int j = 0; j < 2; ++j)
#pragma unroll
                for (int mt = 0; mt < 2; ++mt) {
                    int ti = (g * 2 + j) * 2 + mt;
                    acc[g][j][mt] = *(const g_cv8f*)(ip + ((size_t)(wave * 16 + ti) * 32 + lane) * 8);
                }
        gates_mma(acc, WdecG, hbuf, xbuf, 8, 0, wave, lane);
        __syncthreads();
        lstm_update(acc, c, hbuf, wave, lane);
        __syncthreads();

        // y = h_new @ W_lin^T + b_lin : 14 tiles (2 M x 7 N) over 8 waves
        const g_cbf16* Wl = WlinG;
        OPAQUE_S(Wl);                 // no hoisting of W_lin fragments either
        for (int ti = wave; ti < 2 * NT_Y; ti += 8) {
            int mt = ti / NT_Y, nyt = ti % NT_Y;
            v8f y = (v8f)0.f;
            for (int kt = 0; kt < 8; ++kt) {
                v16bf a = ld_a_lds(hbuf, mt * 16, kt * 32, H, lane);
                v16bf b = ld_b_pk(Wl, kt * NT_Y + nyt, lane);
                y = wmma_bf16(a, b, y);
            }
            int ny = nyt * 16 + (lane & 15);
            if (ny < NY) {
                float bl = b_lin[ny];
                int mb = m0 + mt * 16 + ((lane >> 4) << 3);
#pragma unroll
                for (int e = 0; e < 8; ++e)
                    out[(size_t)(mb + e) * (NDEC * NY) + s * NY + ny] = y[e] + bl;
            }
        }
    }
}

// ---------------- host launcher ----------------
extern "C" void kernel_launch(void* const* d_in, const int* in_sizes, int n_in,
                              void* d_out, int out_size, void* d_ws, size_t ws_size,
                              hipStream_t stream) {
    const float* x     = (const float*)d_in[0];
    const float* W_ih  = (const float*)d_in[1];
    const float* W_hh  = (const float*)d_in[2];
    const float* b_ih  = (const float*)d_in[3];
    const float* b_hh  = (const float*)d_in[4];
    const float* Wc_ih = (const float*)d_in[5];
    const float* Wc_hh = (const float*)d_in[6];
    const float* bc_ih = (const float*)d_in[7];
    const float* bc_hh = (const float*)d_in[8];
    const float* W_lin = (const float*)d_in[9];
    const float* b_lin = (const float*)d_in[10];
    float* out = (float*)d_out;

    bf16*  wsb      = (bf16*)d_ws;
    bf16*  Wenc     = wsb + WENC_ELE;
    bf16*  Wdec     = wsb + WDEC_ELE;
    bf16*  Wcin     = wsb + WCIN_ELE;
    bf16*  Wlin     = wsb + WLIN_ELE;
    float* bias_dec = (float*)((char*)d_ws + BIAS_BYTE);
    float* iproj    = (float*)((char*)d_ws + IPROJ_BYTE);

    pack_enc <<<(320 * G4 + 255) / 256, 256, 0, stream>>>(W_ih, W_hh, b_ih, b_hh, Wenc);
    pack_h256<<<(H * G4 + 255) / 256, 256, 0, stream>>>(Wc_hh, Wdec);
    pack_h256<<<(H * G4 + 255) / 256, 256, 0, stream>>>(Wc_ih, Wcin);
    pack_lin <<<(H * NT_Y * 16 + 255) / 256, 256, 0, stream>>>(W_lin, Wlin);
    pack_bias<<<(G4 + 255) / 256, 256, 0, stream>>>(bc_ih, bc_hh, bias_dec);

    lstm_fused<<<NWG, 256, 0, stream>>>(x, Wenc, Wdec, Wcin, Wlin,
                                        bias_dec, b_lin, out, iproj);
}
